// DilatedAttentionOP_28140625723860
// MI455X (gfx1250) — matrix operand order repaired and checked by reading, not compiled, not run on hardware
//
#include <hip/hip_runtime.h>

// ---------------------------------------------------------------------------
// Dilated attention (B=2, L=8192, D=1024, H=16, hd=64, SEG=512)
// pairs = [(0,0),(1,0),(1,1)]  ->  out = 1/3*att0(x) + 1/3*att1(parity view)
// Compute-bound GEMMs -> bf16 WMMA (v_wmma_f32_16x16x32_bf16) everywhere.
// GEMM stage loads use gfx1250 async-to-LDS DMA with double buffering.
// ---------------------------------------------------------------------------

typedef __bf16 bf16_t;
typedef __attribute__((ext_vector_type(16))) __bf16 v16bf;
typedef __attribute__((ext_vector_type(8)))  __bf16 v8bf;
typedef __attribute__((ext_vector_type(4)))  __bf16 v4bf;
typedef __attribute__((ext_vector_type(8)))  float  v8f;

#define WMMA_BF16(a, b, c) \
  __builtin_amdgcn_wmma_f32_16x16x32_bf16(false, (a), false, (b), (short)0, (c), false, false)

static constexpr int Bc  = 2;
static constexpr int Lc  = 8192;
static constexpr int Dc  = 1024;
static constexpr int Hc  = 16;
static constexpr int HDc = 64;
static constexpr int Sc  = 512;
static constexpr int Mtok = Bc * Lc;       // 16384 tokens per branch
static constexpr int N3  = 3 * Dc;         // 3072

__device__ __forceinline__ v16bf concat8(v8bf lo, v8bf hi) {
  return __builtin_shufflevector(lo, hi, 0,1,2,3,4,5,6,7,8,9,10,11,12,13,14,15);
}

// A-matrix fragment, 16x32 bf16 (ISA 7.12.2): lane&15 = row M; lanes<16 hold
// K {0..7,16..23}, lanes>=16 hold K {8..15,24..31}.
__device__ __forceinline__ v16bf frag_a(const bf16_t* p, int stride) {
  const int lane = threadIdx.x & 31;
  const bf16_t* q = p + (lane & 15) * stride + ((lane >> 4) << 3);
  v8bf lo = *(const v8bf*)q;
  v8bf hi = *(const v8bf*)(q + 16);
  return concat8(lo, hi);
}

// B-matrix fragment, 32x16 bf16: lane&15 = column N; lanes<16 hold
// K {0..7,16..23}, lanes>=16 hold K {8..15,24..31}. p points at [n][k] storage.
__device__ __forceinline__ v16bf frag_b(const bf16_t* p, int stride) {
  const int lane = threadIdx.x & 31;
  const bf16_t* q = p + (lane & 15) * stride + ((lane >> 4) << 3);
  v8bf lo = *(const v8bf*)q;
  v8bf hi = *(const v8bf*)(q + 16);
  return concat8(lo, hi);
}

// gfx1250 async DMA: 16 bytes global -> LDS, tracked by ASYNCcnt.
// vdst = LDS byte address (generic-pointer low 32 bits), vaddr = 64b global.
__device__ __forceinline__ void async_copy16(unsigned lds, unsigned long long gaddr) {
  asm volatile("global_load_async_to_lds_b128 %0, %1, off"
               :: "v"(lds), "v"(gaddr) : "memory");
}
__device__ __forceinline__ void wait_async0() {
  asm volatile("s_wait_asynccnt 0x0" ::: "memory");
}

// ---------------------------------------------------------------------------
// x (f32, [B,L,D]) -> xh0 bf16 natural order; xh1 bf16 parity-gathered order
// (per batch: even l first, then odd l) so 512-chunks == dilated segments.
// ---------------------------------------------------------------------------
__global__ void k_cast_gather(const float* __restrict__ x,
                              bf16_t* __restrict__ xh0,
                              bf16_t* __restrict__ xh1) {
  size_t idx = ((size_t)blockIdx.x * blockDim.x + threadIdx.x) * 4;
  if (idx >= (size_t)Mtok * Dc) return;
  const float4 v = *(const float4*)(x + idx);
  v4bf o;
  o[0] = (bf16_t)v.x; o[1] = (bf16_t)v.y; o[2] = (bf16_t)v.z; o[3] = (bf16_t)v.w;
  *(v4bf*)(xh0 + idx) = o;
  size_t d = idx % Dc;
  size_t t = idx / Dc;
  size_t l = t % Lc, b = t / Lc;
  size_t j = (l & 1) ? (Lc / 2 + (l >> 1)) : (l >> 1);
  *(v4bf*)(xh1 + (b * Lc + j) * Dc + d) = o;
}

__global__ void k_cast(const float* __restrict__ src, bf16_t* __restrict__ dst, int n) {
  int idx = (blockIdx.x * blockDim.x + threadIdx.x) * 4;
  if (idx >= n) return;
  const float4 v = *(const float4*)(src + idx);
  v4bf o;
  o[0] = (bf16_t)v.x; o[1] = (bf16_t)v.y; o[2] = (bf16_t)v.z; o[3] = (bf16_t)v.w;
  *(v4bf*)(dst + idx) = o;
}

// ---------------------------------------------------------------------------
// C[M,N] = A[M,K] @ W[N,K]^T + bias.  128x128x32 tiles, 8 waves/block, each
// wave a 32x64 subtile (2x4 wmma tiles). Double-buffered LDS tiles filled by
// async-to-LDS DMA overlapped with the WMMA stream; last k-tile peeled so the
// steady-state loop has no staging predicate.
//   mode 0: store bf16 to Cout
//   mode 1: Cout f32  = (acc+bias)/3                (branch-0 out-proj)
//   mode 2: Cout f32 += (acc+bias)/3 at parity-permuted row (branch-1)
// ---------------------------------------------------------------------------
#define BM 128
#define BN 128
#define BK 32
#define LDT 40  // padded LDS row stride (elements); 80B keeps 16B alignment

__global__ __launch_bounds__(256) void k_gemm(const bf16_t* __restrict__ A,
                                              const bf16_t* __restrict__ W,
                                              const float* __restrict__ bias,
                                              void* __restrict__ Cout,
                                              int M, int N, int K, int mode) {
  __shared__ __attribute__((aligned(16))) bf16_t As[2 * BM * LDT];
  __shared__ __attribute__((aligned(16))) bf16_t Ws[2 * BN * LDT];
  const int tid = threadIdx.x, lane = tid & 31, wid = tid >> 5;
  const int bm = blockIdx.y * BM, bn = blockIdx.x * BN;
  const int wm = (wid >> 1) * 32, wn = (wid & 1) * 64;
  const int ldrow = tid >> 1, ldoff = (tid & 1) * 16;

  v8f zero = {0.f, 0.f, 0.f, 0.f, 0.f, 0.f, 0.f, 0.f};
  v8f acc[2][4];
#pragma unroll
  for (int mi = 0; mi < 2; ++mi)
#pragma unroll
    for (int ni = 0; ni < 4; ++ni) acc[mi][ni] = zero;

  const bf16_t* gA = A + (size_t)(bm + ldrow) * K + ldoff;
  const bf16_t* gW = W + (size_t)(bn + ldrow) * K + ldoff;
  const unsigned asAddr = (unsigned)(unsigned long long)(As + ldrow * LDT + ldoff);
  const unsigned wsAddr = (unsigned)(unsigned long long)(Ws + ldrow * LDT + ldoff);

  auto stage = [&](int buf, int k0) {
    const unsigned ao = asAddr + (unsigned)buf * (BM * LDT * 2);
    const unsigned wo = wsAddr + (unsigned)buf * (BN * LDT * 2);
    async_copy16(ao,      (unsigned long long)(gA + k0));
    async_copy16(ao + 16, (unsigned long long)(gA + k0 + 8));
    async_copy16(wo,      (unsigned long long)(gW + k0));
    async_copy16(wo + 16, (unsigned long long)(gW + k0 + 8));
  };
  auto compute = [&](int buf) {
    const bf16_t* Ab = As + buf * (BM * LDT);
    const bf16_t* Wb = Ws + buf * (BN * LDT);
    v16bf af0 = frag_a(Ab + (wm + 0)  * LDT, LDT);
    v16bf af1 = frag_a(Ab + (wm + 16) * LDT, LDT);
#pragma unroll
    for (int ni = 0; ni < 4; ++ni) {
      v16bf bfr = frag_b(Wb + (wn + ni * 16) * LDT, LDT);
      acc[0][ni] = WMMA_BF16(af0, bfr, acc[0][ni]);
      acc[1][ni] = WMMA_BF16(af1, bfr, acc[1][ni]);
    }
  };

  const int nk = K / BK;
  stage(0, 0);
  int cur = 0;
  for (int t = 0; t + 1 < nk; ++t, cur ^= 1) {
    wait_async0();        // this wave's async fills of buf[cur] are done
    __syncthreads();      // ... and every other wave's too
    __builtin_prefetch(gA + (t + 2) * BK, 0, 0);   // global_prefetch_b8
    stage(cur ^ 1, (t + 1) * BK);                  // overlap next tile fill
    compute(cur);
  }
  wait_async0();          // tail tile: nothing left to stage
  __syncthreads();
  compute(cur);

  const float wgt = 1.0f / 3.0f;
#pragma unroll
  for (int mi = 0; mi < 2; ++mi)
#pragma unroll
    for (int ni = 0; ni < 4; ++ni) {
      v8f a = acc[mi][ni];
      int col = bn + wn + ni * 16 + (lane & 15);
      float bv = bias[col];
#pragma unroll
      for (int i = 0; i < 8; ++i) {
        int row = bm + wm + mi * 16 + i + ((lane >> 4) << 3);
        float v = a[i] + bv;
        if (mode == 0) {
          ((bf16_t*)Cout)[(size_t)row * N + col] = (bf16_t)v;
        } else if (mode == 1) {
          ((float*)Cout)[(size_t)row * N + col] = v * wgt;
        } else {
          int b = row >> 13, j = row & (Lc - 1);
          int l = (j < Lc / 2) ? (j << 1) : (((j - Lc / 2) << 1) | 1);
          ((float*)Cout)[((size_t)b * Lc + l) * N + col] += v * wgt;
        }
      }
    }
}

// ---------------------------------------------------------------------------
// Segment attention: one block = 64 query rows (4 waves x 16) of one
// (segment, head); streams 32-key tiles with online softmax.
// qkv layout per token: [q0..1023 | k0..1023 | v0..1023] bf16.
// ---------------------------------------------------------------------------
__global__ __launch_bounds__(128) void k_attn(const bf16_t* __restrict__ qkv,
                                              bf16_t* __restrict__ Oout) {
  __shared__ __attribute__((aligned(16))) bf16_t Vs[HDc * LDT];       // [d][k]
  __shared__ __attribute__((aligned(16))) bf16_t Ps[4 * 16 * LDT];    // per-wave P
  const int tid = threadIdx.x, lane = tid & 31, wid = tid >> 5;
  const int h = blockIdx.y;
  const int base = blockIdx.z * Sc;                 // segment token base
  const int qrow0 = blockIdx.x * 64 + wid * 16;
  const float scale = 0.125f;                       // 1/sqrt(64)

  const bf16_t* qbase = qkv + (size_t)(base + qrow0) * N3 + h * HDc;
  v16bf qf0 = frag_a(qbase, N3);                    // d = 0..31
  v16bf qf1 = frag_a(qbase + 32, N3);               // d = 32..63

  v8f zero = {0.f, 0.f, 0.f, 0.f, 0.f, 0.f, 0.f, 0.f};
  v8f o[4];
  float m_[8], l_[8];
#pragma unroll
  for (int i = 0; i < 8; ++i) { m_[i] = -1e30f; l_[i] = 0.f; }
#pragma unroll
  for (int j = 0; j < 4; ++j) o[j] = zero;

  for (int kt = 0; kt < Sc; kt += 32) {
    // cooperative transpose-stage of V tile: Vs[d][k]
    {
      int kk = tid >> 2;                    // 0..31 key within tile
      int d0 = (tid & 3) * 16;
      const bf16_t* vp = qkv + (size_t)(base + kt + kk) * N3 + 2 * Dc + h * HDc + d0;
      v8bf va = *(const v8bf*)vp;
      v8bf vb = *(const v8bf*)(vp + 8);
#pragma unroll
      for (int q = 0; q < 8; ++q) {
        Vs[(d0 + q) * LDT + kk]     = va[q];
        Vs[(d0 + 8 + q) * LDT + kk] = vb[q];
      }
    }
    __syncthreads();

    // scores: S[16 x 32] = Q(16x64) . K^T, two 16x16 column chunks
    v8f sc[2];
#pragma unroll
    for (int c = 0; c < 2; ++c) {
      const bf16_t* kp = qkv + (size_t)(base + kt + c * 16) * N3 + Dc + h * HDc;
      v16bf kf0 = frag_b(kp, N3);           // d 0..31
      v16bf kf1 = frag_b(kp + 32, N3);      // d 32..63
      v8f z = zero;
      z = WMMA_BF16(qf0, kf0, z);
      z = WMMA_BF16(qf1, kf1, z);
      sc[c] = z;
    }

    // online softmax (row lives across the 16-lane half; reduce with xor 1..8)
    float p0[8], p1[8], alpha[8];
#pragma unroll
    for (int i = 0; i < 8; ++i) {
      float s0 = sc[0][i] * scale, s1 = sc[1][i] * scale;
      float mx = fmaxf(s0, s1);
#pragma unroll
      for (int msk = 1; msk < 16; msk <<= 1) mx = fmaxf(mx, __shfl_xor(mx, msk, 32));
      float nm = fmaxf(m_[i], mx);
      alpha[i] = __expf(m_[i] - nm);
      m_[i] = nm;
      p0[i] = __expf(s0 - nm);
      p1[i] = __expf(s1 - nm);
      float rs = p0[i] + p1[i];
#pragma unroll
      for (int msk = 1; msk < 16; msk <<= 1) rs += __shfl_xor(rs, msk, 32);
      l_[i] = l_[i] * alpha[i] + rs;
    }
#pragma unroll
    for (int j = 0; j < 4; ++j)
#pragma unroll
      for (int i = 0; i < 8; ++i) o[j][i] *= alpha[i];

    // C-layout P -> LDS -> A-layout fragment (per-wave buffer; LDS in-order)
    bf16_t* ps = Ps + wid * 16 * LDT;
    const int rowb = (lane >> 4) * 8, coll = lane & 15;
#pragma unroll
    for (int i = 0; i < 8; ++i) {
      ps[(rowb + i) * LDT + coll]      = (bf16_t)p0[i];
      ps[(rowb + i) * LDT + 16 + coll] = (bf16_t)p1[i];
    }
    v16bf pf = frag_a(ps, LDT);
#pragma unroll
    for (int j = 0; j < 4; ++j) {
      v16bf vf = frag_b(Vs + (j * 16) * LDT, LDT);
      o[j] = WMMA_BF16(pf, vf, o[j]);
    }
    __syncthreads();
  }

#pragma unroll
  for (int j = 0; j < 4; ++j)
#pragma unroll
    for (int i = 0; i < 8; ++i) {
      int row = qrow0 + i + ((lane >> 4) << 3);
      int col = j * 16 + (lane & 15);
      Oout[(size_t)(base + row) * Dc + h * HDc + col] = (bf16_t)(o[j][i] / l_[i]);
    }
}

// ---------------------------------------------------------------------------
extern "C" void kernel_launch(void* const* d_in, const int* in_sizes, int n_in,
                              void* d_out, int out_size, void* d_ws, size_t ws_size,
                              hipStream_t stream) {
  const float* x    = (const float*)d_in[0];
  const float* Wqkv = (const float*)d_in[1];
  const float* bqkv = (const float*)d_in[2];
  const float* Wout = (const float*)d_in[3];
  const float* bout = (const float*)d_in[4];
  float* out = (float*)d_out;

  char* ws = (char*)d_ws;
  auto alloc = [&](size_t bytes) -> char* {
    char* p = ws;
    ws += (bytes + 255) & ~(size_t)255;
    return p;
  };
  bf16_t* xh0    = (bf16_t*)alloc((size_t)Mtok * Dc * 2);
  bf16_t* xh1    = (bf16_t*)alloc((size_t)Mtok * Dc * 2);
  bf16_t* WhQKV  = (bf16_t*)alloc((size_t)2 * N3 * Dc * 2);
  bf16_t* WhOut  = (bf16_t*)alloc((size_t)2 * Dc * Dc * 2);
  bf16_t* qkv0   = (bf16_t*)alloc((size_t)Mtok * N3 * 2);
  bf16_t* qkv1   = (bf16_t*)alloc((size_t)Mtok * N3 * 2);
  bf16_t* O0     = (bf16_t*)alloc((size_t)Mtok * Dc * 2);
  bf16_t* O1     = (bf16_t*)alloc((size_t)Mtok * Dc * 2);

  // 1) cast + dilated gather
  {
    int threads = Mtok * Dc / 4;
    k_cast_gather<<<(threads + 255) / 256, 256, 0, stream>>>(x, xh0, xh1);
  }
  // 2) cast weights (branches 0,1 are the leading contiguous slabs)
  {
    int n = 2 * N3 * Dc;
    k_cast<<<(n / 4 + 255) / 256, 256, 0, stream>>>(Wqkv, WhQKV, n);
    n = 2 * Dc * Dc;
    k_cast<<<(n / 4 + 255) / 256, 256, 0, stream>>>(Wout, WhOut, n);
  }
  // 3) QKV projections
  k_gemm<<<dim3(N3 / BN, Mtok / BM), 256, 0, stream>>>(xh0, WhQKV, bqkv, qkv0,
                                                       Mtok, N3, Dc, 0);
  k_gemm<<<dim3(N3 / BN, Mtok / BM), 256, 0, stream>>>(xh1, WhQKV + (size_t)N3 * Dc,
                                                       bqkv + N3, qkv1, Mtok, N3, Dc, 0);
  // 4) segment attention (32 segments x 16 heads x 8 q-blocks)
  k_attn<<<dim3(Sc / 64, Hc, Mtok / Sc), 128, 0, stream>>>(qkv0, O0);
  k_attn<<<dim3(Sc / 64, Hc, Mtok / Sc), 128, 0, stream>>>(qkv1, O1);
  // 5) output projections -> d_out (branch0 overwrites, branch1 scatter-adds)
  k_gemm<<<dim3(Dc / BN, Mtok / BM), 256, 0, stream>>>(O0, WhOut, bout, out,
                                                       Mtok, Dc, Dc, 1);
  k_gemm<<<dim3(Dc / BN, Mtok / BM), 256, 0, stream>>>(O1, WhOut + (size_t)Dc * Dc,
                                                       bout + Dc, out, Mtok, Dc, Dc, 2);
}
